// TemporalAttention_41197326303781
// MI455X (gfx1250) — matrix operand order repaired and checked by reading, not compiled, not run on hardware
//
#include <hip/hip_runtime.h>
#include <hip/hip_bf16.h>

#define D_MODEL 1024
#define N_HEAD  16
#define SEQ     1024
#define BATCH   4
#define DK      64
#define MTOT    (BATCH * SEQ)   // 4096 rows

#define LDS_PITCH 80            // bytes per 32-element bf16 row (16B aligned, conflict-spread)

// Async global->LDS path (GLOBAL_LOAD_ASYNC_TO_LDS_B128 + s_wait_asynccnt),
// guarded so a toolchain without the builtin falls back to reg staging.
#if defined(__has_builtin)
#  if __has_builtin(__builtin_amdgcn_global_load_async_to_lds_b128) && \
      __has_builtin(__builtin_amdgcn_s_wait_asynccnt)
#    define USE_ASYNC_LDS 1
#  endif
#endif
#ifndef USE_ASYNC_LDS
#  define USE_ASYNC_LDS 0
#endif

#if USE_ASYNC_LDS
// builtin expects: (int4(vector_size 16) AS1*, int4 AS3*, imm int, imm int)
typedef int i4g __attribute__((__vector_size__(16)));
typedef __attribute__((address_space(1))) void* as1vp;
typedef __attribute__((address_space(3))) void* as3vp;
typedef __attribute__((address_space(1))) i4g*  as1p;
typedef __attribute__((address_space(3))) i4g*  as3p;
#define GPTR16(p) ((as1p)(as1vp)(void*)(p))
#define LPTR16(p) ((as3p)(as3vp)(void*)(p))
#endif

typedef __attribute__((ext_vector_type(16))) __bf16 v16bf;
typedef __attribute__((ext_vector_type(8)))  float  v8f;

union Frag {
    v16bf v;
    uint4 q[2];
    unsigned int u[8];
};

__device__ __forceinline__ unsigned short f2bf(float f) {
    unsigned int u = __builtin_bit_cast(unsigned int, f);
    u += 0x7fffu + ((u >> 16) & 1u);               // round-to-nearest-even
    return (unsigned short)(u >> 16);
}

// ---------------------------------------------------------------------------
// 1) x + positional encoding; keep f32 residual + bf16 copy for WMMA GEMMs
// ---------------------------------------------------------------------------
__global__ __launch_bounds__(256) void pe_add_kernel(
    const float* __restrict__ x, float* __restrict__ residual,
    unsigned short* __restrict__ xbf) {
    int idx = blockIdx.x * 256 + threadIdx.x;
    int d = idx & (D_MODEL - 1);
    int s = (idx >> 10) & (SEQ - 1);
    float deven = (float)(d & ~1);
    float freq = expf(deven * (-9.210340371976184f / (float)D_MODEL));
    float ang  = (float)s * freq;
    float pe   = (d & 1) ? cosf(ang) : sinf(ang);
    float v = x[idx] + pe;
    residual[idx] = v;
    xbf[idx] = f2bf(v);
}

// ---------------------------------------------------------------------------
// 2) cast all four weight matrices f32 -> bf16
// ---------------------------------------------------------------------------
__global__ __launch_bounds__(256) void wconv_kernel(
    const float* __restrict__ Wq, const float* __restrict__ Wk,
    const float* __restrict__ Wv, const float* __restrict__ Wo,
    unsigned short* __restrict__ wqb, unsigned short* __restrict__ wkb,
    unsigned short* __restrict__ wvb, unsigned short* __restrict__ wob) {
    int idx = blockIdx.x * 256 + threadIdx.x;
    wqb[idx] = f2bf(Wq[idx]);
    wkb[idx] = f2bf(Wk[idx]);
    wvb[idx] = f2bf(Wv[idx]);
    wob[idx] = f2bf(Wo[idx]);
}

// ---------------------------------------------------------------------------
// 3) C[M=4096, N=1024] = A @ W^T, bf16 WMMA, K = 1024.
//    mode 0: bf16 out, row-major [M, D]                (Q, K projections)
//    mode 2: bf16 out, per-head transposed [B,H,DK,S]  (V projection)
//    mode 3: f32 out, row-major                        (output projection)
//    Block = 8 waves; wave w owns a 64x16 strip (4 accum tiles along M).
//    The 64x32 A tile is shared by all 8 waves -> double-buffered in LDS,
//    staged with async global->LDS DMA when available.
// ---------------------------------------------------------------------------
__global__ __launch_bounds__(256) void gemm_xw_kernel(
    const unsigned short* __restrict__ A, const unsigned short* __restrict__ W,
    unsigned short* __restrict__ outb, float* __restrict__ outf,
    int mode, float scale) {
    const int K = D_MODEL;
    __shared__ alignas(16) unsigned char ldsA[2][64 * LDS_PITCH];

    int tid  = threadIdx.x;
    int wave = tid >> 5;
    int lane = tid & 31;
    int lane16 = lane & 15;
    int hi = (lane >> 4) & 1;

    int mbase = blockIdx.x * 64;
    int ncol  = blockIdx.y * 128 + wave * 16 + lane16;

    const unsigned short* wrow = W + (size_t)ncol * K + (hi ? 16 : 0);

    // staging assignment: thread -> (row 0..63, 16B chunk 0..3) of the A tile
    int srow   = tid >> 2;
    int schunk = tid & 3;
    const unsigned short* gsrc = A + (size_t)(mbase + srow) * K + schunk * 8;
    unsigned char* sdst0 = &ldsA[0][srow * LDS_PITCH + schunk * 16];
    unsigned char* sdst1 = &ldsA[1][srow * LDS_PITCH + schunk * 16];

    // prologue: stage tile for kk = 0
#if USE_ASYNC_LDS
    __builtin_amdgcn_global_load_async_to_lds_b128(GPTR16(gsrc), LPTR16(sdst0), 0, 0);
    __builtin_amdgcn_s_wait_asynccnt(0);
#else
    *(uint4*)sdst0 = *(const uint4*)(gsrc);
#endif
    __syncthreads();

    v8f acc[4] = {};
    for (int kk = 0; kk < K; kk += 32) {
        int buf = (kk >> 5) & 1;
        bool have_next = (kk + 32) < K;
#if USE_ASYNC_LDS
        if (have_next)
            __builtin_amdgcn_global_load_async_to_lds_b128(
                GPTR16(gsrc + kk + 32), LPTR16(buf ? sdst0 : sdst1), 0, 0);
#else
        uint4 stg;
        if (have_next) stg = *(const uint4*)(gsrc + kk + 32);  // in flight over WMMAs
#endif
        __builtin_prefetch(wrow + kk + 288, 0, 3);             // global_prefetch_b8

        Frag bfr;
        bfr.q[0] = *(const uint4*)(wrow + kk);
        bfr.q[1] = *(const uint4*)(wrow + kk + 8);

        const unsigned char* lb = ldsA[buf];
        Frag a[4];
#pragma unroll
        for (int t = 0; t < 4; ++t) {
            const unsigned char* ap = lb + (size_t)(t * 16 + lane16) * LDS_PITCH + hi * 32;
            a[t].q[0] = *(const uint4*)(ap);
            a[t].q[1] = *(const uint4*)(ap + 16);
        }
#pragma unroll
        for (int t = 0; t < 4; ++t)
            acc[t] = __builtin_amdgcn_wmma_f32_16x16x32_bf16(
                false, a[t].v, false, bfr.v, (short)0, acc[t], false, false);

#if USE_ASYNC_LDS
        if (have_next) __builtin_amdgcn_s_wait_asynccnt(0);
#else
        if (have_next) *(uint4*)(buf ? sdst0 : sdst1) = stg;
#endif
        __syncthreads();
    }

#pragma unroll
    for (int t = 0; t < 4; ++t) {
#pragma unroll
        for (int r = 0; r < 8; ++r) {
            int m = mbase + t * 16 + r + hi * 8;
            float v = acc[t][r] * scale;
            if (mode == 0) {
                outb[(size_t)m * D_MODEL + ncol] = f2bf(v);
            } else if (mode == 2) {
                int b_ = m >> 10, s = m & (SEQ - 1);
                int h = ncol >> 6, d = ncol & (DK - 1);
                outb[((size_t)(b_ * N_HEAD + h) * DK + d) * SEQ + s] = f2bf(v);
            } else {
                outf[(size_t)m * D_MODEL + ncol] = v;
            }
        }
    }
}

// ---------------------------------------------------------------------------
// 4) scores[b,h,q,k] = (Q.K)/8 per head: M=N=1024, Kdim=64 (2 WMMA steps).
//    All fragment loads are hoisted ahead of the 8 WMMAs.
//    Writes raw (pre-softmax) scores as f32 directly into the attn output.
// ---------------------------------------------------------------------------
__global__ __launch_bounds__(256) void qk_kernel(
    const unsigned short* __restrict__ Q, const unsigned short* __restrict__ Km,
    float* __restrict__ scores) {
    int wave = threadIdx.x >> 5;
    int lane = threadIdx.x & 31;
    int lane16 = lane & 15;
    int hi = (lane >> 4) & 1;
    int bh = blockIdx.z;
    int b = bh >> 4, h = bh & 15;

    const unsigned short* Qh = Q  + (size_t)b * SEQ * D_MODEL + h * DK;
    const unsigned short* Kh = Km + (size_t)b * SEQ * D_MODEL + h * DK;
    float* Sp = scores + (size_t)bh * SEQ * SEQ;

    int mbase = blockIdx.x * 64;
    int ncol  = blockIdx.y * 128 + wave * 16 + lane16;

    const unsigned short* krow  = Kh + (size_t)ncol * D_MODEL + (hi ? 16 : 0);
    const unsigned short* qrow0 = Qh + (size_t)(mbase + lane16) * D_MODEL + (hi ? 16 : 0);

    Frag bf0, bf1, a0[4], a1[4];
    bf0.q[0] = *(const uint4*)(krow);
    bf0.q[1] = *(const uint4*)(krow + 8);
    bf1.q[0] = *(const uint4*)(krow + 32);
    bf1.q[1] = *(const uint4*)(krow + 40);
#pragma unroll
    for (int t = 0; t < 4; ++t) {
        const unsigned short* ap = qrow0 + (size_t)(t * 16) * D_MODEL;
        a0[t].q[0] = *(const uint4*)(ap);
        a0[t].q[1] = *(const uint4*)(ap + 8);
        a1[t].q[0] = *(const uint4*)(ap + 32);
        a1[t].q[1] = *(const uint4*)(ap + 40);
    }

    v8f acc[4] = {};
#pragma unroll
    for (int t = 0; t < 4; ++t)
        acc[t] = __builtin_amdgcn_wmma_f32_16x16x32_bf16(
            false, a0[t].v, false, bf0.v, (short)0, acc[t], false, false);
#pragma unroll
    for (int t = 0; t < 4; ++t)
        acc[t] = __builtin_amdgcn_wmma_f32_16x16x32_bf16(
            false, a1[t].v, false, bf1.v, (short)0, acc[t], false, false);

#pragma unroll
    for (int t = 0; t < 4; ++t) {
#pragma unroll
        for (int r = 0; r < 8; ++r) {
            int m = mbase + t * 16 + r + hi * 8;
            Sp[(size_t)m * SEQ + ncol] = acc[t][r] * 0.125f;  // 1/sqrt(64)
        }
    }
}

// ---------------------------------------------------------------------------
// 5) in-place softmax over each row of attn (one block per row, 1024 elems)
// ---------------------------------------------------------------------------
__global__ __launch_bounds__(256) void softmax_kernel(float* __restrict__ attn) {
    __shared__ float red[256];
    int tid = threadIdx.x;
    float4* row = (float4*)(attn + (size_t)blockIdx.x * SEQ);
    float4 v = row[tid];
    float m = fmaxf(fmaxf(v.x, v.y), fmaxf(v.z, v.w));
    red[tid] = m; __syncthreads();
    for (int s = 128; s > 0; s >>= 1) {
        if (tid < s) red[tid] = fmaxf(red[tid], red[tid + s]);
        __syncthreads();
    }
    float rowmax = red[0]; __syncthreads();
    v.x = __expf(v.x - rowmax); v.y = __expf(v.y - rowmax);
    v.z = __expf(v.z - rowmax); v.w = __expf(v.w - rowmax);
    red[tid] = v.x + v.y + v.z + v.w; __syncthreads();
    for (int s = 128; s > 0; s >>= 1) {
        if (tid < s) red[tid] += red[tid + s];
        __syncthreads();
    }
    float inv = 1.0f / red[0];
    v.x *= inv; v.y *= inv; v.z *= inv; v.w *= inv;
    row[tid] = v;
}

// ---------------------------------------------------------------------------
// 6) ctx = attn @ V per head: M=1024, N=64, Kdim=1024.
//    The 64x32 V tile is shared by all 8 waves -> double-buffered in LDS
//    (async DMA when available).  Per-wave f32 attn rows are register
//    double-buffered + converted to bf16.
// ---------------------------------------------------------------------------
__global__ __launch_bounds__(256) void av_kernel(
    const float* __restrict__ attn, const unsigned short* __restrict__ vT,
    unsigned short* __restrict__ ctx) {
    __shared__ alignas(16) unsigned char ldsV[2][64 * LDS_PITCH];

    int tid  = threadIdx.x;
    int wave = tid >> 5;
    int lane = tid & 31;
    int lane16 = lane & 15;
    int hi = (lane >> 4) & 1;
    int bh = blockIdx.y;
    int b = bh >> 4, h = bh & 15;

    const float* Ah = attn + (size_t)bh * SEQ * SEQ;
    const unsigned short* Vh = vT + (size_t)bh * DK * SEQ;

    int mrow = blockIdx.x * 128 + wave * 16 + lane16;
    const float* arow = Ah + (size_t)mrow * SEQ + (hi ? 8 : 0);

    // staging assignment: thread -> (d row 0..63, 16B chunk 0..3) of the V tile
    int srow   = tid >> 2;
    int schunk = tid & 3;
    const unsigned short* gsrcV = Vh + (size_t)srow * SEQ + schunk * 8;
    unsigned char* sdst0 = &ldsV[0][srow * LDS_PITCH + schunk * 16];
    unsigned char* sdst1 = &ldsV[1][srow * LDS_PITCH + schunk * 16];

    // prologue: stage V tile for kk = 0; preload attn rows for kk = 0
#if USE_ASYNC_LDS
    __builtin_amdgcn_global_load_async_to_lds_b128(GPTR16(gsrcV), LPTR16(sdst0), 0, 0);
#else
    *(uint4*)sdst0 = *(const uint4*)(gsrcV);
#endif
    float4 f[4];
    f[0] = ((const float4*)(arow))[0];
    f[1] = ((const float4*)(arow))[1];
    f[2] = ((const float4*)(arow + 16))[0];
    f[3] = ((const float4*)(arow + 16))[1];
#if USE_ASYNC_LDS
    __builtin_amdgcn_s_wait_asynccnt(0);
#endif
    __syncthreads();

    v8f acc[4] = {};
    for (int kk = 0; kk < SEQ; kk += 32) {
        int buf = (kk >> 5) & 1;
        bool have_next = (kk + 32) < SEQ;
        float4 g[4];
#if USE_ASYNC_LDS
        if (have_next)
            __builtin_amdgcn_global_load_async_to_lds_b128(
                GPTR16(gsrcV + kk + 32), LPTR16(buf ? sdst0 : sdst1), 0, 0);
#else
        uint4 stg;
        if (have_next) stg = *(const uint4*)(gsrcV + kk + 32);
#endif
        if (have_next) {
            const int kn = kk + 32;
            g[0] = ((const float4*)(arow + kn))[0];
            g[1] = ((const float4*)(arow + kn))[1];
            g[2] = ((const float4*)(arow + kn + 16))[0];
            g[3] = ((const float4*)(arow + kn + 16))[1];
        }

        // convert current attn chunk to a bf16 A fragment
        Frag a;
#pragma unroll
        for (int j = 0; j < 4; ++j) {
            a.u[2 * j]     = (unsigned int)f2bf(f[j].x) | ((unsigned int)f2bf(f[j].y) << 16);
            a.u[2 * j + 1] = (unsigned int)f2bf(f[j].z) | ((unsigned int)f2bf(f[j].w) << 16);
        }

        const unsigned char* lb = ldsV[buf];
#pragma unroll
        for (int t = 0; t < 4; ++t) {
            Frag bt;
            const unsigned char* bp = lb + (size_t)(t * 16 + lane16) * LDS_PITCH + hi * 16;
            bt.q[0] = *(const uint4*)(bp);
            bt.q[1] = *(const uint4*)(bp + 32);
            acc[t] = __builtin_amdgcn_wmma_f32_16x16x32_bf16(
                false, a.v, false, bt.v, (short)0, acc[t], false, false);
        }

        if (have_next) {
#if USE_ASYNC_LDS
            __builtin_amdgcn_s_wait_asynccnt(0);
#else
            *(uint4*)(buf ? sdst0 : sdst1) = stg;
#endif
#pragma unroll
            for (int j = 0; j < 4; ++j) f[j] = g[j];
        }
        __syncthreads();
    }

#pragma unroll
    for (int t = 0; t < 4; ++t) {
#pragma unroll
        for (int r = 0; r < 8; ++r) {
            int m = blockIdx.x * 128 + wave * 16 + r + hi * 8;
            int d = t * 16 + lane16;
            ctx[(size_t)(b * SEQ + m) * D_MODEL + h * DK + d] = f2bf(acc[t][r]);
        }
    }
}

// ---------------------------------------------------------------------------
// 7) y = LayerNorm(proj + residual) * gamma + beta   (one block per row)
// ---------------------------------------------------------------------------
__global__ __launch_bounds__(256) void ln_kernel(
    const float* __restrict__ proj, const float* __restrict__ residual,
    const float* __restrict__ gamma, const float* __restrict__ beta,
    float* __restrict__ y) {
    __shared__ float red[256];
    int tid = threadIdx.x;
    size_t base = (size_t)blockIdx.x * D_MODEL;
    float4 pv = ((const float4*)(proj + base))[tid];
    float4 rv = ((const float4*)(residual + base))[tid];
    float4 v;
    v.x = pv.x + rv.x; v.y = pv.y + rv.y; v.z = pv.z + rv.z; v.w = pv.w + rv.w;

    red[tid] = v.x + v.y + v.z + v.w; __syncthreads();
    for (int s = 128; s > 0; s >>= 1) {
        if (tid < s) red[tid] += red[tid + s];
        __syncthreads();
    }
    float mean = red[0] * (1.0f / D_MODEL); __syncthreads();

    float dx = v.x - mean, dy = v.y - mean, dz = v.z - mean, dw = v.w - mean;
    red[tid] = dx * dx + dy * dy + dz * dz + dw * dw; __syncthreads();
    for (int s = 128; s > 0; s >>= 1) {
        if (tid < s) red[tid] += red[tid + s];
        __syncthreads();
    }
    float rstd = rsqrtf(red[0] * (1.0f / D_MODEL) + 1e-6f);

    float4 g = ((const float4*)gamma)[tid];
    float4 bb = ((const float4*)beta)[tid];
    float4 o;
    o.x = dx * rstd * g.x + bb.x;
    o.y = dy * rstd * g.y + bb.y;
    o.z = dz * rstd * g.z + bb.z;
    o.w = dw * rstd * g.w + bb.w;
    ((float4*)(y + base))[tid] = o;
}

// ---------------------------------------------------------------------------
extern "C" void kernel_launch(void* const* d_in, const int* in_sizes, int n_in,
                              void* d_out, int out_size, void* d_ws, size_t ws_size,
                              hipStream_t stream) {
    const float* x     = (const float*)d_in[0];
    const float* Wq    = (const float*)d_in[1];
    const float* Wk    = (const float*)d_in[2];
    const float* Wv    = (const float*)d_in[3];
    const float* Wo    = (const float*)d_in[4];
    const float* gamma = (const float*)d_in[5];
    const float* beta  = (const float*)d_in[6];

    const size_t MD = (size_t)MTOT * D_MODEL;     // 4 Mi elements
    const size_t WN = (size_t)D_MODEL * D_MODEL;  // 1 Mi elements

    char* ws = (char*)d_ws;
    float*          residual = (float*)ws;          ws += MD * sizeof(float);
    unsigned short* xbf      = (unsigned short*)ws; ws += MD * 2;
    unsigned short* wqb      = (unsigned short*)ws; ws += WN * 2;
    unsigned short* wkb      = (unsigned short*)ws; ws += WN * 2;
    unsigned short* wvb      = (unsigned short*)ws; ws += WN * 2;
    unsigned short* wob      = (unsigned short*)ws; ws += WN * 2;
    unsigned short* qb       = (unsigned short*)ws; ws += MD * 2;
    unsigned short* kb       = (unsigned short*)ws; ws += MD * 2;
    unsigned short* vtb      = (unsigned short*)ws; ws += MD * 2;
    unsigned short* ctxb     = (unsigned short*)ws; ws += MD * 2;
    float*          proj     = (float*)ws;          ws += MD * sizeof(float);

    float* y    = (float*)d_out;
    float* attn = y + MD;   // tuple order: (y, attn)

    pe_add_kernel<<<(unsigned)(MD / 256), 256, 0, stream>>>(x, residual, xbf);
    wconv_kernel<<<(unsigned)(WN / 256), 256, 0, stream>>>(Wq, Wk, Wv, Wo,
                                                           wqb, wkb, wvb, wob);
    dim3 g1(MTOT / 64, D_MODEL / 128);
    gemm_xw_kernel<<<g1, 256, 0, stream>>>(xbf, wqb, qb,  nullptr, 0, 1.0f);
    gemm_xw_kernel<<<g1, 256, 0, stream>>>(xbf, wkb, kb,  nullptr, 0, 1.0f);
    gemm_xw_kernel<<<g1, 256, 0, stream>>>(xbf, wvb, vtb, nullptr, 2, 1.0f);

    dim3 g2(SEQ / 64, SEQ / 128, BATCH * N_HEAD);
    qk_kernel<<<g2, 256, 0, stream>>>(qb, kb, attn);

    softmax_kernel<<<BATCH * N_HEAD * SEQ, 256, 0, stream>>>(attn);

    dim3 g3(SEQ / 128, BATCH * N_HEAD);
    av_kernel<<<g3, 256, 0, stream>>>(attn, vtb, ctxb);

    gemm_xw_kernel<<<g1, 256, 0, stream>>>(ctxb, wob, nullptr, proj, 3, 1.0f);

    ln_kernel<<<MTOT, 256, 0, stream>>>(proj, residual, gamma, beta, y);
}
